// DTMS_18090402251301
// MI455X (gfx1250) — matrix operand order repaired and checked by reading, not compiled, not run on hardware
//
#include <hip/hip_runtime.h>
#include <hip/hip_bf16.h>
#include <math.h>

typedef __attribute__((ext_vector_type(16))) _Float16 v16h;
typedef __attribute__((ext_vector_type(8)))  _Float16 v8h;
typedef __attribute__((ext_vector_type(8)))  float    v8f;
typedef __attribute__((ext_vector_type(4)))  unsigned int v4u;
typedef __attribute__((ext_vector_type(8)))  int          v8i;
typedef __attribute__((ext_vector_type(4)))  int          v4i;

union UH16 { v16h v; v8h h[2]; };
union UF8  { v8f  v; float f[8]; };

#define DEV __device__ __forceinline__

DEV float siluf(float x){ return x / (1.f + __expf(-x)); }
DEV float gelu_tanh(float x){
  const float c0 = 0.7978845608028654f; // sqrt(2/pi)
  return 0.5f * x * (1.f + tanhf(c0 * (x + 0.044715f * x * x * x)));
}
DEV float softplusf(float x){ return (x > 20.f) ? x : log1pf(__expf(x)); }

// ---------------------------------------------------------------------------
// fp32 -> f16 conversion (weights, done once per launch)
// ---------------------------------------------------------------------------
__global__ void k_cvt16(const float* __restrict__ in, _Float16* __restrict__ out, int n){
  int i = blockIdx.x * blockDim.x + threadIdx.x;
  if (i < n) out[i] = (_Float16)in[i];
}

// ---------------------------------------------------------------------------
// LayerNorm over last dim, one wave32 per row, f16 output (WMMA A-operand).
// mode 0: plain rows from s0 (fp32)
// mode 1: rows = s0 + s1 (merge of scan-direction pairs)
// mode 2: channel interleave of s0/s1 (each 64ch), D = 128
// mode 3: width interleave of s0/s1 (B,H,2W,64), D = 64
// ---------------------------------------------------------------------------
__global__ void k_ln(const float* __restrict__ s0, const float* __restrict__ s1,
                     const float* __restrict__ g, const float* __restrict__ bt,
                     _Float16* __restrict__ out,
                     int Mrows, int D, int mode, int H, int W, int L)
{
  int wv  = threadIdx.x >> 5;
  int lid = threadIdx.x & 31;
  int row = blockIdx.x * (blockDim.x >> 5) + wv;
  if (row >= Mrows) return;
  float vals[8];
  int nv = D >> 5;
#pragma unroll
  for (int i = 0; i < 8; i++){
    vals[i] = 0.f;
    if (i < nv){
      int c = lid + (i << 5);
      float v;
      if (mode == 0){
        v = s0[(size_t)row * D + c];
      } else if (mode == 1){
        size_t ix = (size_t)row * D + c;
        v = s0[ix] + s1[ix];
      } else if (mode == 2){
        size_t base = (size_t)row * 64 + (c >> 1);
        v = (c & 1) ? s1[base] : s0[base];
      } else {
        int b  = row / L;
        int l  = row % L;
        int w2 = l % (2 * W);
        int h  = l / (2 * W);
        size_t base = (((size_t)b * H + h) * W + (w2 >> 1)) * 64 + c;
        v = (w2 & 1) ? s1[base] : s0[base];
      }
      vals[i] = v;
    }
  }
  float s = 0.f;
#pragma unroll
  for (int i = 0; i < 8; i++) s += vals[i];
#pragma unroll
  for (int off = 16; off >= 1; off >>= 1) s += __shfl_xor(s, off, 32);
  float mu = s / (float)D;
  float vs = 0.f;
#pragma unroll
  for (int i = 0; i < 8; i++){
    if (i < nv){ float dd = vals[i] - mu; vs += dd * dd; }
  }
#pragma unroll
  for (int off = 16; off >= 1; off >>= 1) vs += __shfl_xor(vs, off, 32);
  float inv = rsqrtf(vs / (float)D + 1e-5f);
#pragma unroll
  for (int i = 0; i < 8; i++){
    if (i < nv){
      int c = lid + (i << 5);
      out[(size_t)row * D + c] = (_Float16)((vals[i] - mu) * inv * g[c] + bt[c]);
    }
  }
}

// ---------------------------------------------------------------------------
// WMMA f16 GEMM: C[M,Nact] = A[M,K] * W[Nact,K]^T, fp32 accumulate.
// Block = 128 threads (4 waves) -> 64 rows x 64 cols of C.
// Wave -> 16 rows x 64 cols (4 accumulators; A fragment reused 4x).
// B tile (64 x chunk) staged in LDS by the Tensor Data Mover
// (tensor_load_to_lds + s_wait_tensorcnt), with TDM pad_enable inserting
// 16B per row so ds_load_b128 fragment reads spread across banks.
// Operand layout per CDNA5 ISA 7.12.2:
//   A: lane m (0..15) holds K[0..7],K[16..23]; lane m+16 holds K[8..15],K[24..31]
//   B: lane n holds W[n][K0..7],W[n][K16..23]; lane n+16 holds W[n][K8..15],W[n][K24..31]
// ---------------------------------------------------------------------------
enum { EPI_F32 = 0, EPI_RES_CH, EPI_RES_W, EPI_BIAS_GELU_H,
       EPI_BIAS_RES_F32, EPI_SB_RELU_F32, EPI_SB_RELU_NCHW };

#define KCHUNK 256
#define LDS_PAD 8   /* halfs = 16 bytes, matches TDM pad_amount of 4 DWORDs */

template<int EPI>
__global__ void __launch_bounds__(128)
k_gemm(const _Float16* __restrict__ A, const _Float16* __restrict__ Wt,
       int M, int Nact, int K,
       float* __restrict__ outF, _Float16* __restrict__ outH,
       const float* __restrict__ bias, const float* __restrict__ scale,
       const float* __restrict__ resA, const float* __restrict__ resB,
       int H, int W, int L)
{
  __shared__ _Float16 ldsW[64 * (KCHUNK + LDS_PAD)];
  int wv  = threadIdx.x >> 5;
  int lid = threadIdx.x & 31;
  int m0  = (blockIdx.x * 4 + wv) * 16;
  int n0  = blockIdx.y * 64;
  int r16 = lid & 15;
  int hs  = lid >> 4;
  const _Float16* aRow = A + (size_t)(m0 + r16) * K;
  v8f acc[4] = {v8f{}, v8f{}, v8f{}, v8f{}};

  for (int kc = 0; kc < K; kc += KCHUNK){
    int chunkH = K - kc; if (chunkH > KCHUNK) chunkH = KCHUNK;
    int ldsStride = chunkH + LDS_PAD;
    __syncthreads();   // previous chunk fully consumed
#if __has_builtin(__builtin_amdgcn_tensor_load_to_lds)
    if (wv == 0){
      // ---- Tensor DMA descriptor (D#), 2D tile: 64 rows x chunkH*2 bytes ----
      unsigned long long ga = (unsigned long long)(const void*)(Wt + (size_t)n0 * K + kc);
      int rows   = Nact - n0;           // OOB rows beyond tensor_dim1 load zero
      int chunkB = chunkH * 2;          // bytes per tile row (data_size = 1B)
      int pi = __builtin_ctz((unsigned)chunkH) - 2;   // pad interval: 2^(pi+1) DWORDs
      v4u g0;
      g0[0] = 1u;                                         // count=1, user mode
      g0[1] = 0u;                                         // lds_addr (ldsW at base)
      g0[2] = (unsigned)(ga & 0xffffffffu);
      g0[3] = (unsigned)((ga >> 32) & 0x01ffffffu) | (2u << 30);  // addr hi | type=2
      v8i g1;
      g1[0] = (1 << 20) | (pi << 22) | (3 << 25);         // pad_enable, interval, 4 DWORDs
      g1[1] = (chunkB & 0xffff) << 16;                    // tensor_dim0[15:0]
      g1[2] = ((chunkB >> 16) & 0xffff) | ((rows & 0xffff) << 16);
      g1[3] = ((rows >> 16) & 0xffff) | ((chunkB & 0xffff) << 16);  // tile_dim0
      g1[4] = 64;                                         // tile_dim1 = 64 rows
      g1[5] = K * 2;                                      // tensor_dim0_stride (bytes)
      g1[6] = 0;
      g1[7] = 0;
      v4i z4 = {0, 0, 0, 0};
#if defined(__clang_major__) && (__clang_major__ >= 23)
      v8i z8 = {0, 0, 0, 0, 0, 0, 0, 0};
      __builtin_amdgcn_tensor_load_to_lds(g0, g1, z4, z4, z8, 0);
#else
      __builtin_amdgcn_tensor_load_to_lds(g0, g1, z4, z4, 0);
#endif
      __builtin_amdgcn_s_wait_tensorcnt(0);
    }
#else
    // fallback: cooperative staging (keeps file compiling without the builtin)
    for (int idx = threadIdx.x; idx < 64 * chunkH; idx += 128){
      int r = idx / chunkH, cix = idx % chunkH;
      int grow = n0 + r;
      ldsW[r * ldsStride + cix] =
          (grow < Nact) ? Wt[(size_t)grow * K + kc + cix] : (_Float16)0.f;
    }
#endif
    __syncthreads();   // LDS tile visible to all waves

    for (int kk = 0; kk < chunkH; kk += 32){
      const _Float16* ap = aRow + kc + kk;
      __builtin_prefetch(ap + 32, 0, 3);
      UH16 a;
      a.h[0] = *(const v8h*)(ap + hs * 8);
      a.h[1] = *(const v8h*)(ap + 16 + hs * 8);
#pragma unroll
      for (int sub = 0; sub < 4; sub++){
        const _Float16* bp = &ldsW[(sub * 16 + r16) * ldsStride + kk + hs * 8];
        UH16 b;
        b.h[0] = *(const v8h*)(bp);
        b.h[1] = *(const v8h*)(bp + 16);
        acc[sub] = __builtin_amdgcn_wmma_f32_16x16x32_f16(false, a.v, false, b.v,
                                                          (short)0, acc[sub], false, false);
      }
    }
  }

  int mb = m0 + hs * 8;
#pragma unroll
  for (int sub = 0; sub < 4; sub++){
    int col = n0 + sub * 16 + r16;
    if (col >= Nact) continue;
    UF8 c; c.v = acc[sub];
#pragma unroll
    for (int r = 0; r < 8; r++){
      int m = mb + r;
      float v = c.f[r];
      if (EPI == EPI_F32){
        outF[(size_t)m * Nact + col] = v;
      } else if (EPI == EPI_RES_CH){
        size_t base = (size_t)m * 64 + (col >> 1);
        float res = (col & 1) ? resB[base] : resA[base];
        outF[(size_t)m * Nact + col] = v + res;
      } else if (EPI == EPI_RES_W){
        int b_ = m / L; int l = m % L;
        int w2 = l % (2 * W); int h = l / (2 * W);
        size_t base = (((size_t)b_ * H + h) * W + (w2 >> 1)) * 64 + col;
        float res = (w2 & 1) ? resB[base] : resA[base];
        outF[(size_t)m * Nact + col] = v + res;
      } else if (EPI == EPI_BIAS_GELU_H){
        v += bias[col];
        outH[(size_t)m * Nact + col] = (_Float16)gelu_tanh(v);
      } else if (EPI == EPI_BIAS_RES_F32){
        v += bias[col] + resA[(size_t)m * Nact + col];
        outF[(size_t)m * Nact + col] = v;
      } else if (EPI == EPI_SB_RELU_F32){
        v = v * scale[col] + bias[col];
        outF[(size_t)m * Nact + col] = v > 0.f ? v : 0.f;
      } else { // EPI_SB_RELU_NCHW
        v = v * scale[col] + bias[col];
        v = v > 0.f ? v : 0.f;
        int b_ = m / (H * W); int hw = m % (H * W);
        outF[((size_t)b_ * Nact + col) * (size_t)(H * W) + hw] = v;
      }
    }
  }
}

// ---------------------------------------------------------------------------
// Depthwise 3x3 conv (NHWC, pad 1) + SiLU; writes fp32 (scan) and f16 (GEMM A).
// ---------------------------------------------------------------------------
__global__ void k_dwconv(const float* __restrict__ U, const float* __restrict__ Wc,
                         float* __restrict__ V, _Float16* __restrict__ Vh,
                         int B, int H, int W, int Di)
{
  size_t i = (size_t)blockIdx.x * blockDim.x + threadIdx.x;
  size_t total = (size_t)B * H * W * Di;
  if (i >= total) return;
  int d = (int)(i % Di);
  size_t t = i / Di;
  int w = (int)(t % W); t /= W;
  int h = (int)(t % H);
  int b = (int)(t / H);
  float acc = 0.f;
  const float* wk = Wc + (size_t)d * 9;
#pragma unroll
  for (int dy = -1; dy <= 1; dy++){
    int hh = h + dy;
    if (hh < 0 || hh >= H) continue;
#pragma unroll
    for (int dx = -1; dx <= 1; dx++){
      int ww = w + dx;
      if (ww < 0 || ww >= W) continue;
      acc += wk[(dy + 1) * 3 + (dx + 1)] * U[(((size_t)b * H + hh) * W + ww) * Di + d];
    }
  }
  float s = siluf(acc);
  V[i]  = s;
  Vh[i] = (_Float16)s;
}

// ---------------------------------------------------------------------------
// Fused selective scan: one block per (b, d, direction-pair).
// pair 0: k=0 fwd + k=2 bwd over row-major order
// pair 1: k=1 fwd + k=3 bwd over column-major order
// dt = softplus(dt_w @ dts + dt_b); a = exp(dt*A); bu = dt*Bs*u; h = a h + bu
// y(pos) += h*Cs + D*u.  Chunked block scan (phase1 local, serial carries,
// phase2 recompute+emit). Both directions write the pair's own buffer, so the
// 4-way cross-merge is a plain elementwise sum done inside the next LN.
// ---------------------------------------------------------------------------
struct DirP { const float* dw; float db; float A; float Dk; int koff; };

DEV void scan_elem(const float* __restrict__ V, const float* __restrict__ XD,
                   size_t rowBase, int lmem, int Di, int d, int XDw, int dr,
                   const DirP& dp, float& a, float& bu, float& Cs, float& u)
{
  u = V[(rowBase + lmem) * (size_t)Di + d];
  const float* xd = XD + (rowBase + lmem) * (size_t)XDw + dp.koff;
  float s = dp.db;
  for (int r = 0; r < dr; r++) s += dp.dw[r] * xd[r];
  float dt = softplusf(s);
  a  = __expf(dt * dp.A);
  bu = dt * xd[dr] * u;
  Cs = xd[dr + 1];
}

DEV int posmap(int p, int pair, int Hh, int Ww){
  return pair ? ((p % Hh) * Ww + p / Hh) : p;
}

__global__ void k_scan(const float* __restrict__ V, const float* __restrict__ XD,
                       const float* __restrict__ dtw, const float* __restrict__ dtb,
                       const float* __restrict__ Alog, const float* __restrict__ Dp,
                       float* __restrict__ yA, float* __restrict__ yB,
                       int B, int Hh, int Ww, int Di, int dr, int L)
{
  int t    = threadIdx.x;
  int blk  = blockIdx.x;
  int pair = blk & 1;
  int rest = blk >> 1;
  int d = rest % Di;
  int b = rest / Di;
  int kF = pair, kB = pair + 2;
  int nc  = dr + 2;
  int XDw = 4 * nc;
  DirP dF, dB;
  dF.dw = dtw + ((size_t)kF * Di + d) * dr;  dB.dw = dtw + ((size_t)kB * Di + d) * dr;
  dF.db = dtb[kF * Di + d];                  dB.db = dtb[kB * Di + d];
  dF.A  = -__expf(Alog[kF * Di + d]);        dB.A  = -__expf(Alog[kB * Di + d]);
  dF.Dk = Dp[kF * Di + d];                   dB.Dk = Dp[kB * Di + d];
  dF.koff = kF * nc;                         dB.koff = kB * nc;
  float* Y = pair ? yB : yA;
  size_t rowBase = (size_t)b * L;
  int Lc = L >> 8;
  int p0 = t * Lc;

  __shared__ float sA[256], sH[256], sA2[256], sG[256], cF[256], cB[256];

  // phase 1: local forward & backward summaries
  float ap = 1.f, hp = 0.f;
  for (int j = 0; j < Lc; j++){
    int lmem = posmap(p0 + j, pair, Hh, Ww);
    float a, bu, Cs, u;
    scan_elem(V, XD, rowBase, lmem, Di, d, XDw, dr, dF, a, bu, Cs, u);
    hp = a * hp + bu; ap *= a;
  }
  float ab = 1.f, gb = 0.f;
  for (int j = Lc - 1; j >= 0; j--){
    int lmem = posmap(p0 + j, pair, Hh, Ww);
    float a, bu, Cs, u;
    scan_elem(V, XD, rowBase, lmem, Di, d, XDw, dr, dB, a, bu, Cs, u);
    gb = a * gb + bu; ab *= a;
  }
  sA[t] = ap; sH[t] = hp; sA2[t] = ab; sG[t] = gb;
  __syncthreads();
  if (t == 0){
    float c = 0.f;
    for (int i = 0; i < 256; i++){ cF[i] = c; c = sA[i] * c + sH[i]; }
    float c2 = 0.f;
    for (int i = 255; i >= 0; i--){ cB[i] = c2; c2 = sA2[i] * c2 + sG[i]; }
  }
  __syncthreads();

  // phase 2: recompute with carries, emit y contributions
  float h = cF[t];
  for (int j = 0; j < Lc; j++){
    int lmem = posmap(p0 + j, pair, Hh, Ww);
    float a, bu, Cs, u;
    scan_elem(V, XD, rowBase, lmem, Di, d, XDw, dr, dF, a, bu, Cs, u);
    h = a * h + bu;
    Y[(rowBase + lmem) * (size_t)Di + d] = h * Cs + dF.Dk * u;
  }
  float g = cB[t];
  for (int j = Lc - 1; j >= 0; j--){
    int lmem = posmap(p0 + j, pair, Hh, Ww);
    float a, bu, Cs, u;
    scan_elem(V, XD, rowBase, lmem, Di, d, XDw, dr, dB, a, bu, Cs, u);
    g = a * g + bu;
    Y[(rowBase + lmem) * (size_t)Di + d] += g * Cs + dB.Dk * u;
  }
}

// ---------------------------------------------------------------------------
// Pack concat([t1, t2[:, :, :W], t2[:, :, W:]]) -> f16 (B*H*W, 256)
// ---------------------------------------------------------------------------
__global__ void k_pack(const float* __restrict__ t1, const float* __restrict__ t2,
                       _Float16* __restrict__ tc, int B, int H, int W)
{
  size_t i = (size_t)blockIdx.x * blockDim.x + threadIdx.x;
  size_t total = (size_t)B * H * W * 256;
  if (i >= total) return;
  int c = (int)(i % 256);
  size_t m = i / 256;
  int w = (int)(m % W); size_t t = m / W;
  int h = (int)(t % H); int b = (int)(t / H);
  float v;
  if (c < 128){
    v = t1[m * 128 + c];
  } else {
    int cc = c - 128;
    int wOff = (cc >= 64) ? (W + w) : w;
    int ch = cc & 63;
    v = t2[(((size_t)b * H + h) * (2 * W) + wOff) * 64 + ch];
  }
  tc[i] = (_Float16)v;
}

// ---------------------------------------------------------------------------
// Dense 3x3 conv 64->64 (NHWC, pad 1) + scale/bias/ReLU, f16 out for last GEMM
// ---------------------------------------------------------------------------
__global__ void k_conv3(const float* __restrict__ X, const float* __restrict__ Wc,
                        const float* __restrict__ sc, const float* __restrict__ bi,
                        _Float16* __restrict__ out, int B, int H, int W)
{
  size_t i = (size_t)blockIdx.x * blockDim.x + threadIdx.x;
  size_t total = (size_t)B * H * W * 64;
  if (i >= total) return;
  int oc = (int)(i & 63);
  size_t m = i >> 6;
  int w = (int)(m % W); size_t t = m / W;
  int h = (int)(t % H); int b = (int)(t / H);
  float acc = 0.f;
  for (int dy = -1; dy <= 1; dy++){
    int hh = h + dy; if (hh < 0 || hh >= H) continue;
    for (int dx = -1; dx <= 1; dx++){
      int ww = w + dx; if (ww < 0 || ww >= W) continue;
      const float* xin  = X + (((size_t)b * H + hh) * W + ww) * 64;
      const float* wrow = Wc + (size_t)oc * 576 + ((dy + 1) * 3 + (dx + 1));
      for (int ic = 0; ic < 64; ic++)
        acc += wrow[ic * 9] * xin[ic];
    }
  }
  acc = acc * sc[oc] + bi[oc];
  out[i] = (_Float16)(acc > 0.f ? acc : 0.f);
}

// ---------------------------------------------------------------------------
// Launcher
// ---------------------------------------------------------------------------
extern "C" void kernel_launch(void* const* d_in, const int* in_sizes, int n_in,
                              void* d_out, int out_size, void* d_ws, size_t ws_size,
                              hipStream_t stream)
{
  (void)in_sizes; (void)n_in; (void)out_size; (void)ws_size;
  const float* x1 = (const float*)d_in[0];
  const float* x2 = (const float*)d_in[1];
  const float* S1[18]; for (int i = 0; i < 18; i++) S1[i] = (const float*)d_in[2 + i];
  const float* S2[18]; for (int i = 0; i < 18; i++) S2[i] = (const float*)d_in[20 + i];
  const float* w1 = (const float*)d_in[38];
  const float* s1 = (const float*)d_in[39];
  const float* b1 = (const float*)d_in[40];
  const float* w2 = (const float*)d_in[41];
  const float* s2 = (const float*)d_in[42];
  const float* b2 = (const float*)d_in[43];
  const float* w3 = (const float*)d_in[44];
  const float* s3 = (const float*)d_in[45];
  const float* b3 = (const float*)d_in[46];

  size_t off = 0;
  auto alloc = [&](size_t bytes) -> void* {
    void* p = (char*)d_ws + off;
    off += (bytes + 255) & ~(size_t)255;
    return p;
  };
  // scratch regions (max over both branches; aggressively aliased)
  float*    R_U    = (float*)alloc((size_t)16 << 20);    // U -> yA -> tcat
  float*    R_V    = (float*)alloc((size_t)16 << 20);    // V -> C1
  _Float16* R_Vh   = (_Float16*)alloc((size_t)8 << 20);  // Vh -> C2h
  float*    R_XD   = (float*)alloc((size_t)4 << 20);
  float*    R_yB   = (float*)alloc((size_t)16 << 20);    // yB -> Gh (f16)
  _Float16* R_ln1h = (_Float16*)alloc((size_t)4 << 20);
  _Float16* R_zh   = (_Float16*)alloc((size_t)8 << 20);
  float*    R_xres = (float*)alloc((size_t)8 << 20);
  _Float16* R_ln2h = (_Float16*)alloc((size_t)4 << 20);
  float*    R_t1   = (float*)alloc((size_t)8 << 20);
  float*    R_t2   = (float*)alloc((size_t)8 << 20);

  auto cvt = [&](const float* src, size_t n) -> _Float16* {
    _Float16* dst = (_Float16*)alloc(n * 2);
    k_cvt16<<<dim3((unsigned)((n + 255) / 256)), 256, 0, stream>>>(src, dst, (int)n);
    return dst;
  };
  _Float16* inproj1h  = cvt(S1[2],  (size_t)256 * 128);
  _Float16* xproj1h   = cvt(S1[4],  (size_t)40 * 256);
  _Float16* outproj1h = cvt(S1[11], (size_t)128 * 256);
  _Float16* fc11h     = cvt(S1[14], (size_t)512 * 128);
  _Float16* fc21h     = cvt(S1[16], (size_t)128 * 512);
  _Float16* inproj2h  = cvt(S2[2],  (size_t)128 * 64);
  _Float16* xproj2h   = cvt(S2[4],  (size_t)24 * 128);
  _Float16* outproj2h = cvt(S2[11], (size_t)64 * 128);
  _Float16* fc12h     = cvt(S2[14], (size_t)256 * 64);
  _Float16* fc22h     = cvt(S2[16], (size_t)64 * 256);
  _Float16* w1h       = cvt(w1,     (size_t)64 * 256);
  _Float16* w3h       = cvt(w3,     (size_t)64 * 64);

  auto nblk = [](int n){ return (unsigned)((n + 63) / 64); };

  auto runBranch = [&](int mode, int Hh, int Wg, int dmod, int Di, int dr,
                       const float* const* P,
                       _Float16* inpH, _Float16* xprH, _Float16* outpH,
                       _Float16* f1H, _Float16* f2H, float* tOut)
  {
    const int Bn = 4;
    int L = Hh * Wg;
    int M = Bn * L;
    int Nxd = 4 * (dr + 2);
    // 1. interleave + LN1 -> ln1h (f16)
    k_ln<<<dim3(M / 4), 128, 0, stream>>>(x1, x2, P[0], P[1], R_ln1h, M, dmod, mode, 64, 64, L);
    // 2. in_proj GEMM (WMMA+TDM) -> U (fp32)
    k_gemm<EPI_F32><<<dim3(M / 64, nblk(Di)), 128, 0, stream>>>(
        R_ln1h, inpH, M, Di, dmod, R_U, nullptr, nullptr, nullptr, nullptr, nullptr, 64, 64, L);
    // 3. depthwise conv 3x3 + SiLU -> V, Vh
    {
      size_t tot = (size_t)M * Di;
      k_dwconv<<<dim3((unsigned)((tot + 255) / 256)), 256, 0, stream>>>(R_U, P[3], R_V, R_Vh, Bn, Hh, Wg, Di);
    }
    // 4. x_proj GEMM (WMMA, N padded) -> XD
    k_gemm<EPI_F32><<<dim3(M / 64, nblk(Nxd)), 128, 0, stream>>>(
        R_Vh, xprH, M, Nxd, Di, R_XD, nullptr, nullptr, nullptr, nullptr, nullptr, 64, 64, L);
    // 5. fused selective scan, 4 directions -> yA (=R_U), yB
    k_scan<<<dim3(Bn * Di * 2), 256, 0, stream>>>(
        R_V, R_XD, P[5], P[6], P[7], P[8], R_U, R_yB, Bn, Hh, Wg, Di, dr, L);
    // 6. cross-merge + inner LN -> zh (f16)
    k_ln<<<dim3(M / 4), 128, 0, stream>>>(R_U, R_yB, P[9], P[10], R_zh, M, Di, 1, 64, 64, L);
    // 7. out_proj GEMM + interleave residual -> xres
    if (mode == 2)
      k_gemm<EPI_RES_CH><<<dim3(M / 64, nblk(dmod)), 128, 0, stream>>>(
          R_zh, outpH, M, dmod, Di, R_xres, nullptr, nullptr, nullptr, x1, x2, 64, 64, L);
    else
      k_gemm<EPI_RES_W><<<dim3(M / 64, nblk(dmod)), 128, 0, stream>>>(
          R_zh, outpH, M, dmod, Di, R_xres, nullptr, nullptr, nullptr, x1, x2, 64, 64, L);
    // 8. LN2 -> ln2h
    k_ln<<<dim3(M / 4), 128, 0, stream>>>(R_xres, nullptr, P[12], P[13], R_ln2h, M, dmod, 0, 64, 64, L);
    // 9. fc1 GEMM + bias + GELU -> Gh (f16, aliases yB)
    _Float16* Gh = (_Float16*)R_yB;
    k_gemm<EPI_BIAS_GELU_H><<<dim3(M / 64, nblk(4 * dmod)), 128, 0, stream>>>(
        R_ln2h, f1H, M, 4 * dmod, dmod, nullptr, Gh, P[15], nullptr, nullptr, nullptr, 64, 64, L);
    // 10. fc2 GEMM + bias + residual -> tOut
    k_gemm<EPI_BIAS_RES_F32><<<dim3(M / 64, nblk(dmod)), 128, 0, stream>>>(
        Gh, f2H, M, dmod, 4 * dmod, tOut, nullptr, P[17], nullptr, R_xres, nullptr, 64, 64, L);
  };

  // branch 1: channel interleave, d_model=128, Di=256, dr=8, grid 64x64
  runBranch(2, 64, 64, 128, 256, 8, S1, inproj1h, xproj1h, outproj1h, fc11h, fc21h, R_t1);
  // branch 2: width interleave, d_model=64, Di=128, dr=4, grid 64x128
  runBranch(3, 64, 128, 64, 128, 4, S2, inproj2h, xproj2h, outproj2h, fc12h, fc22h, R_t2);

  // CBR head
  _Float16* tcat = (_Float16*)R_U;
  {
    size_t tot = (size_t)4 * 64 * 64 * 256;
    k_pack<<<dim3((unsigned)((tot + 255) / 256)), 256, 0, stream>>>(R_t1, R_t2, tcat, 4, 64, 64);
  }
  float* C1 = R_V;
  k_gemm<EPI_SB_RELU_F32><<<dim3(16384 / 64, 1), 128, 0, stream>>>(
      tcat, w1h, 16384, 64, 256, C1, nullptr, b1, s1, nullptr, nullptr, 64, 64, 4096);
  _Float16* C2h = R_Vh;
  {
    size_t tot = (size_t)4 * 64 * 64 * 64;
    k_conv3<<<dim3((unsigned)((tot + 255) / 256)), 256, 0, stream>>>(C1, w2, s2, b2, C2h, 4, 64, 64);
  }
  k_gemm<EPI_SB_RELU_NCHW><<<dim3(16384 / 64, 1), 128, 0, stream>>>(
      C2h, w3h, 16384, 64, 64, (float*)d_out, nullptr, b3, s3, nullptr, nullptr, 64, 64, 4096);
}
